// BasicUpdateBlock_54597624266892
// MI455X (gfx1250) — matrix operand order repaired and checked by reading, not compiled, not run on hardware
//
#include <hip/hip_runtime.h>

// ---------------------------------------------------------------------------
// Types for CDNA5 WMMA (wave32, v_wmma_f32_16x16x32_bf16)
// ---------------------------------------------------------------------------
typedef __bf16 bf16_t;
typedef bf16_t bf16x8  __attribute__((ext_vector_type(8)));
typedef bf16_t bf16x16 __attribute__((ext_vector_type(16)));
typedef float  v8f     __attribute__((ext_vector_type(8)));

#define H_    64
#define W_    128
#define HW_   8192        // H*W
#define NTOT  65536       // B*H*W  (B=8)
#define BM    128         // block tile M
#define BN    128         // block tile N (== W -> coalesced im2col rows)
#define KT    32          // K step per WMMA stage

// activation codes
#define ACT_NONE 0
#define ACT_RELU 1
#define ACT_SIG  2
#define ACT_TANH 3

// ---------------------------------------------------------------------------
// Weight prep: OIHW fp32 -> zero-padded [M_pad, K_pad] bf16 row-major A-matrix
// ---------------------------------------------------------------------------
__global__ void k_wprep(const float* __restrict__ w, bf16_t* __restrict__ dst,
                        int M, int K, int K_pad, int total)
{
    int i = blockIdx.x * 256 + threadIdx.x;
    if (i >= total) return;
    int m = i / K_pad;
    int k = i - m * K_pad;
    float v = (m < M && k < K) ? w[(size_t)m * K + k] : 0.0f;
    dst[i] = (bf16_t)v;
}

// flat fp32 -> bf16 conversion (graph inputs)
__global__ void k_cvt(const float* __restrict__ s, bf16_t* __restrict__ d, int total)
{
    int i = blockIdx.x * 256 + threadIdx.x;
    if (i >= total) return;
    d[i] = (bf16_t)s[i];
}

// ---------------------------------------------------------------------------
// Implicit-GEMM conv, stride 1, symmetric pad, NCHW(bf16), fused bias+act.
// 256 threads = 8 wave32; wave grid 4(M) x 2(N); wave tile 32M x 64N
// => 8 v_wmma per K-step, double-buffered LDS, 1 barrier per K-step.
// Fragment layouts per CDNA5 ISA:
//   A (16x32):  lane m=0..15 (lane-half picks K-half), elems {g*8.., 16+g*8..}
//   B (32x16):  lane n, elems {g*8.., 16+g*8..} (same K order as A) (N-major LDS)
//   C (16x16):  elem r -> M = r + 8*(lane/16), N = lane%16
// ---------------------------------------------------------------------------
__global__ __launch_bounds__(256) void k_conv_wmma(
    const bf16_t* __restrict__ src, int CinS, int cin0,
    const bf16_t* __restrict__ wA, int M_pad, int K_pad, int Ktot,
    const float* __restrict__ bias, int Cout,
    bf16_t* __restrict__ dstb, float* __restrict__ dstf, int CoutS, int cout0,
    int KH, int KW, int PH, int PW, int act, float scale)
{
    __shared__ __align__(16) bf16_t As[2][BM][KT];
    __shared__ __align__(16) bf16_t Bs[2][BN][KT];

    const int tid    = threadIdx.x;
    const int n0     = blockIdx.x * BN;
    const int m0     = blockIdx.y * BM;
    const int wid    = tid >> 5;
    const int lane   = tid & 31;
    const int g      = lane >> 4;         // K-half / M-half selector
    const int lm     = lane & 15;
    const int wave_m = (wid >> 1) * 32;   // 4 M-waves
    const int wave_n = (wid & 1) * 64;    // 2 N-waves

    const int KHW = KH * KW;

    // --- B staging mapping: thread -> (n_local, 16 consecutive k) ---
    const int nl = tid & 127;
    const int kb = (tid >> 7) * 16;
    const int n  = n0 + nl;
    const int x  = n & (W_ - 1);
    const int y  = (n >> 7) & (H_ - 1);
    const int bb = n >> 13;
    const int sp_xy = (y << 7) + x;
    const bf16_t* __restrict__ srcb = src + ((size_t)bb * CinS + cin0) * HW_;

    // --- A staging mapping: thread -> (m_local, 16 consecutive k) ---
    const int am = tid & 127;
    const int ac = (tid >> 7) << 4;       // 0 or 16

    v8f acc[2][4];
    #pragma unroll
    for (int mi = 0; mi < 2; ++mi)
        #pragma unroll
        for (int t = 0; t < 4; ++t)
            #pragma unroll
            for (int r = 0; r < 8; ++r) acc[mi][t][r] = 0.0f;

    const int nIter = K_pad / KT;

    auto stage = [&](int it, int bufi) {
        const int k0 = it * KT;
        // ---- A tile: two vectorized 16B global loads ----
        bf16x8 av0, av1;
        #pragma unroll
        for (int j = 0; j < 8; ++j) { av0[j] = (bf16_t)0.0f; av1[j] = (bf16_t)0.0f; }
        if (m0 + am < M_pad) {
            const bf16_t* pa = wA + (size_t)(m0 + am) * K_pad + (k0 + ac);
            av0 = *(const bf16x8*)pa;
            av1 = *(const bf16x8*)(pa + 8);
        }
        *(bf16x8*)(&As[bufi][am][ac])     = av0;
        *(bf16x8*)(&As[bufi][am][ac + 8]) = av1;

        // ---- B tile: on-the-fly im2col (bf16 gather, no convert) ----
        int k = k0 + kb;
        if (KHW == 1) {                   // 1x1 conv fast path (no pad, no decode)
            #pragma unroll
            for (int i = 0; i < 16; ++i) {
                bf16_t v = (bf16_t)0.0f;
                if (k + i < Ktot) v = srcb[(size_t)(k + i) * HW_ + sp_xy];
                Bs[bufi][nl][kb + i] = v;
            }
        } else {
            int ci = k / KHW;
            int rem = k - ci * KHW;
            int ky = rem / KW;
            int kx = rem - ky * KW;
            #pragma unroll
            for (int i = 0; i < 16; ++i) {
                bf16_t v = (bf16_t)0.0f;
                if (k + i < Ktot) {
                    int yy = y + ky - PH;
                    int xx = x + kx - PW;
                    if ((unsigned)yy < (unsigned)H_ && (unsigned)xx < (unsigned)W_)
                        v = srcb[(size_t)ci * HW_ + (yy << 7) + xx];
                }
                Bs[bufi][nl][kb + i] = v;
                if (++kx == KW) { kx = 0; if (++ky == KH) { ky = 0; ++ci; } }
            }
        }
    };

    stage(0, 0);

    for (int it = 0; it < nIter; ++it) {
        __syncthreads();
        if (it + 1 < nIter) {
            // prefetch weight tile two K-steps ahead (global_prefetch_b8)
            if (it + 2 < nIter && m0 + am < M_pad)
                __builtin_prefetch(wA + (size_t)(m0 + am) * K_pad + (it + 2) * KT + ac, 0, 1);
            stage(it + 1, (it + 1) & 1);
        }
        const int bufi = it & 1;

        // ---- A fragments (2 M-subtiles) ----
        bf16x16 a[2];
        #pragma unroll
        for (int mi = 0; mi < 2; ++mi) {
            const int row = wave_m + mi * 16 + lm;
            bf16x8 alo = *(const bf16x8*)&As[bufi][row][g * 8];
            bf16x8 ahi = *(const bf16x8*)&As[bufi][row][16 + g * 8];
            a[mi] = __builtin_shufflevector(alo, ahi,
                        0,1,2,3,4,5,6,7,8,9,10,11,12,13,14,15);
        }
        // ---- B fragments, each reused by both A fragments: 8 WMMAs ----
        #pragma unroll
        for (int t = 0; t < 4; ++t) {
            const int bn = wave_n + t * 16 + lm;
            bf16x8 blo = *(const bf16x8*)&Bs[bufi][bn][g * 8];
            bf16x8 bhi = *(const bf16x8*)&Bs[bufi][bn][16 + g * 8];
            bf16x16 bf = __builtin_shufflevector(blo, bhi,
                            0,1,2,3,4,5,6,7,8,9,10,11,12,13,14,15);
            acc[0][t] = __builtin_amdgcn_wmma_f32_16x16x32_bf16(
                            false, a[0], false, bf, (short)0, acc[0][t], false, false);
            acc[1][t] = __builtin_amdgcn_wmma_f32_16x16x32_bf16(
                            false, a[1], false, bf, (short)0, acc[1][t], false, false);
        }
    }

    // ---- epilogue: bias + scale + activation, NCHW scatter ----
    const bool toF32 = (dstf != nullptr);
    #pragma unroll
    for (int mi = 0; mi < 2; ++mi) {
        #pragma unroll
        for (int t = 0; t < 4; ++t) {
            int nn  = n0 + wave_n + t * 16 + lm;
            int ob  = nn >> 13;
            int osp = nn & (HW_ - 1);
            #pragma unroll
            for (int r = 0; r < 8; ++r) {
                int c = m0 + wave_m + mi * 16 + g * 8 + r;
                if (c < Cout) {
                    float v = acc[mi][t][r] + bias[c];
                    v *= scale;
                    if (act == ACT_RELU)      v = fmaxf(v, 0.0f);
                    else if (act == ACT_SIG)  v = 1.0f / (1.0f + __expf(-v));
                    else if (act == ACT_TANH) v = tanhf(v);
                    size_t oi = ((size_t)ob * CoutS + cout0 + c) * HW_ + osp;
                    if (toF32) dstf[oi] = v;
                    else       dstb[oi] = (bf16_t)v;
                }
            }
        }
    }
}

// ---------------------------------------------------------------------------
// Elementwise helpers (channel-sliced NCHW, bf16 activations)
// ---------------------------------------------------------------------------
__global__ void k_copy_f2b(const float* __restrict__ s, int sCS, int sc0,
                           bf16_t* __restrict__ d, int dCS, int dc0,
                           int C, int total)
{
    int i = blockIdx.x * 256 + threadIdx.x;
    if (i >= total) return;
    int sp = i & (HW_ - 1);
    int t  = i >> 13;
    int c  = t % C;
    int b  = t / C;
    d[((size_t)b * dCS + dc0 + c) * HW_ + sp] =
        (bf16_t)s[((size_t)b * sCS + sc0 + c) * HW_ + sp];
}

__global__ void k_copy_b2b(const bf16_t* __restrict__ s, int sCS, int sc0,
                           bf16_t* __restrict__ d, int dCS, int dc0,
                           int C, int total)
{
    int i = blockIdx.x * 256 + threadIdx.x;
    if (i >= total) return;
    int sp = i & (HW_ - 1);
    int t  = i >> 13;
    int c  = t % C;
    int b  = t / C;
    d[((size_t)b * dCS + dc0 + c) * HW_ + sp] =
        s[((size_t)b * sCS + sc0 + c) * HW_ + sp];
}

__global__ void k_copy_b2f(const bf16_t* __restrict__ s, int sCS, int sc0,
                           float* __restrict__ d, int dCS, int dc0,
                           int C, int total)
{
    int i = blockIdx.x * 256 + threadIdx.x;
    if (i >= total) return;
    int sp = i & (HW_ - 1);
    int t  = i >> 13;
    int c  = t % C;
    int b  = t / C;
    d[((size_t)b * dCS + dc0 + c) * HW_ + sp] =
        (float)s[((size_t)b * sCS + sc0 + c) * HW_ + sp];
}

// out = r * h   (128-channel slices)
__global__ void k_rh(const bf16_t* __restrict__ r, int rCS, int rc0,
                     const bf16_t* __restrict__ h, int hCS, int hc0,
                     bf16_t* __restrict__ o, int oCS, int oc0, int total)
{
    int i = blockIdx.x * 256 + threadIdx.x;
    if (i >= total) return;
    int sp = i & (HW_ - 1);
    int t  = i >> 13;
    int c  = t & 127;
    int b  = t >> 7;
    float rv = (float)r[((size_t)b * rCS + rc0 + c) * HW_ + sp];
    float hv = (float)h[((size_t)b * hCS + hc0 + c) * HW_ + sp];
    o[((size_t)b * oCS + oc0 + c) * HW_ + sp] = (bf16_t)(rv * hv);
}

// h = (1-z)*h + z*q   (in place on h slice)
__global__ void k_upd(const bf16_t* __restrict__ z, int zCS, int zc0,
                      const bf16_t* __restrict__ q, int qCS, int qc0,
                      bf16_t* __restrict__ h, int hCS, int hc0, int total)
{
    int i = blockIdx.x * 256 + threadIdx.x;
    if (i >= total) return;
    int sp = i & (HW_ - 1);
    int t  = i >> 13;
    int c  = t & 127;
    int b  = t >> 7;
    float zv = (float)z[((size_t)b * zCS + zc0 + c) * HW_ + sp];
    float qv = (float)q[((size_t)b * qCS + qc0 + c) * HW_ + sp];
    size_t hi = ((size_t)b * hCS + hc0 + c) * HW_ + sp;
    float hv = (float)h[hi];
    h[hi] = (bf16_t)((1.0f - zv) * hv + zv * qv);
}

// ---------------------------------------------------------------------------
// Host orchestration
// ---------------------------------------------------------------------------
extern "C" void kernel_launch(void* const* d_in, const int* in_sizes, int n_in,
                              void* d_out, int out_size, void* d_ws, size_t ws_size,
                              hipStream_t stream)
{
    (void)in_sizes; (void)n_in; (void)out_size; (void)ws_size;

    const float* net  = (const float*)d_in[0];
    const float* inp  = (const float*)d_in[1];
    const float* corr = (const float*)d_in[2];
    const float* flow = (const float*)d_in[3];

    struct LD { int wi, Cout, Cin, KH, KW, PH, PW; };
    static const LD L[15] = {
        { 4, 256, 324, 1, 1, 0, 0},   // convc1
        { 6, 192, 256, 3, 3, 1, 1},   // convc2
        { 8, 128,   2, 7, 7, 3, 3},   // convf1
        {10,  64, 128, 3, 3, 1, 1},   // convf2
        {12, 126, 256, 3, 3, 1, 1},   // conv
        {14, 128, 384, 1, 5, 0, 2},   // convz1
        {16, 128, 384, 1, 5, 0, 2},   // convr1
        {18, 128, 384, 1, 5, 0, 2},   // convq1
        {20, 128, 384, 5, 1, 2, 0},   // convz2
        {22, 128, 384, 5, 1, 2, 0},   // convr2
        {24, 128, 384, 5, 1, 2, 0},   // convq2
        {26, 256, 128, 3, 3, 1, 1},   // fh1
        {28,   2, 256, 3, 3, 1, 1},   // fh2
        {30, 256, 128, 3, 3, 1, 1},   // mask1
        {32, 576, 256, 1, 1, 0, 0},   // mask2
    };

    size_t wofs[15]; int Mp[15], Kp[15];
    size_t wtot = 0;
    for (int i = 0; i < 15; ++i) {
        int K = L[i].Cin * L[i].KH * L[i].KW;
        Kp[i] = (K + 31) & ~31;
        Mp[i] = (L[i].Cout + 15) & ~15;
        wofs[i] = wtot;
        wtot += (size_t)Mp[i] * Kp[i];
    }

    bf16_t* wpool = (bf16_t*)d_ws;
    size_t wbytes = wtot * sizeof(bf16_t);
    bf16_t* bpool = (bf16_t*)((char*)d_ws + ((wbytes + 255) & ~(size_t)255));

    // bf16 activation pools (B*HW = NTOT elements per channel)
    bf16_t* Cb = bpool;                        // 324ch: corr (bf16)
    bf16_t* Fb = Cb + (size_t)324 * NTOT;      //   2ch: flow (bf16)
    bf16_t* F0 = Fb + (size_t)2   * NTOT;      // 256ch: cor1 -> fh/mask tmp
    bf16_t* F1 = F0 + (size_t)256 * NTOT;      // 256ch: [cor2|flo2] -> [r|q]
    bf16_t* F2 = F1 + (size_t)256 * NTOT;      // 128ch: flo1 -> z
    bf16_t* F4 = F2 + (size_t)128 * NTOT;      // 384ch: hx = [h | inp | out | flow]
    bf16_t* F5 = F4 + (size_t)384 * NTOT;      // 384ch: rhx = [r*h | x]

    float* outH    = (float*)d_out;                      // [8,128,64,128]
    float* outMask = outH + (size_t)128 * NTOT;          // [8,576,64,128]
    float* outDF   = outMask + (size_t)576 * NTOT;       // [8,2,64,128]

    // ---- weight prep + input conversion ----
    for (int i = 0; i < 15; ++i) {
        int K = L[i].Cin * L[i].KH * L[i].KW;
        int total = Mp[i] * Kp[i];
        k_wprep<<<(total + 255) / 256, 256, 0, stream>>>(
            (const float*)d_in[L[i].wi], wpool + wofs[i], L[i].Cout, K, Kp[i], total);
    }
    {
        int tc = 324 * NTOT;
        k_cvt<<<(tc + 255) / 256, 256, 0, stream>>>(corr, Cb, tc);
        int tf = 2 * NTOT;
        k_cvt<<<(tf + 255) / 256, 256, 0, stream>>>(flow, Fb, tf);
    }

    auto conv = [&](int li, const bf16_t* s, int CinS, int cin0,
                    bf16_t* db, float* df, int CoutS, int cout0, int act, float scale) {
        const LD& l = L[li];
        dim3 grid(NTOT / BN, (Mp[li] + BM - 1) / BM);
        k_conv_wmma<<<grid, 256, 0, stream>>>(
            s, CinS, cin0, wpool + wofs[li], Mp[li], Kp[li],
            l.Cin * l.KH * l.KW, (const float*)d_in[l.wi + 1], l.Cout,
            db, df, CoutS, cout0, l.KH, l.KW, l.PH, l.PW, act, scale);
    };
    auto copyF2B = [&](const float* s, int sCS, int sc0,
                       bf16_t* d, int dCS, int dc0, int C) {
        int total = C * NTOT;
        k_copy_f2b<<<(total + 255) / 256, 256, 0, stream>>>(s, sCS, sc0, d, dCS, dc0, C, total);
    };
    auto copyB2B = [&](const bf16_t* s, int sCS, int sc0,
                       bf16_t* d, int dCS, int dc0, int C) {
        int total = C * NTOT;
        k_copy_b2b<<<(total + 255) / 256, 256, 0, stream>>>(s, sCS, sc0, d, dCS, dc0, C, total);
    };
    const int T128 = 128 * NTOT;
    auto rh  = [&]() { k_rh <<<(T128 + 255) / 256, 256, 0, stream>>>(F1, 256, 0,   F4, 384, 0,   F5, 384, 0, T128); };
    auto upd = [&]() { k_upd<<<(T128 + 255) / 256, 256, 0, stream>>>(F2, 128, 0,   F1, 256, 128, F4, 384, 0, T128); };

    // ---- BasicMotionEncoder ----
    conv(0, Cb, 324, 0, F0, nullptr, 256, 0,   ACT_RELU, 1.0f);   // cor1
    conv(1, F0, 256, 0, F1, nullptr, 256, 0,   ACT_RELU, 1.0f);   // cor2  -> F1[0:192)
    conv(2, Fb,   2, 0, F2, nullptr, 128, 0,   ACT_RELU, 1.0f);   // flo1
    conv(3, F2, 128, 0, F1, nullptr, 256, 192, ACT_RELU, 1.0f);   // flo2  -> F1[192:256)
    conv(4, F1, 256, 0, F4, nullptr, 384, 256, ACT_RELU, 1.0f);   // out   -> hx[256:382)

    // assemble hx = [h | inp | out | flow], and rhx x-part
    copyF2B(net, 128, 0, F4, 384, 0,   128);   // h
    copyF2B(inp, 128, 0, F4, 384, 128, 128);   // x.inp
    copyB2B(Fb,    2, 0, F4, 384, 382,   2);   // x.flow tail
    copyB2B(F4,  384, 128, F5, 384, 128, 256); // rhx x-part (static)

    // ---- SepConvGRU horizontal (1x5) ----
    conv(5, F4, 384, 0, F2, nullptr, 128, 0,   ACT_SIG,  1.0f);   // z
    conv(6, F4, 384, 0, F1, nullptr, 256, 0,   ACT_SIG,  1.0f);   // r
    rh();
    conv(7, F5, 384, 0, F1, nullptr, 256, 128, ACT_TANH, 1.0f);   // q
    upd();

    // ---- SepConvGRU vertical (5x1) ----
    conv(8,  F4, 384, 0, F2, nullptr, 128, 0,   ACT_SIG,  1.0f);
    conv(9,  F4, 384, 0, F1, nullptr, 256, 0,   ACT_SIG,  1.0f);
    rh();
    conv(10, F5, 384, 0, F1, nullptr, 256, 128, ACT_TANH, 1.0f);
    upd();

    // ---- outputs ----
    {
        int total = 128 * NTOT;
        k_copy_b2f<<<(total + 255) / 256, 256, 0, stream>>>(F4, 384, 0, outH, 128, 0, 128, total); // h
    }
    conv(11, F4, 384, 0, F0, nullptr, 256, 0, ACT_RELU, 1.0f);       // fh1
    conv(12, F0, 256, 0, nullptr, outDF, 2, 0, ACT_NONE, 1.0f);      // fh2 -> delta_flow (fp32)
    conv(13, F4, 384, 0, F0, nullptr, 256, 0, ACT_RELU, 1.0f);       // mask1
    conv(14, F0, 256, 0, nullptr, outMask, 576, 0, ACT_NONE, 0.25f); // mask2 -> mask (fp32)
}
